// GinAgent_17746804867822
// MI455X (gfx1250) — matrix operand order repaired and checked by reading, not compiled, not run on hardware
//
// GinAgent forward for MI455X (gfx1250, wave32, WMMA).
// Edge scorer (dominant: 1.04M edges x (96x64 + 64x32 + 32x1)) runs on
// v_wmma_f32_16x16x32_bf16 (bf16 in, f32 accumulate). BatchNorm batch stats
// computed exactly via 3-pass recompute (stats1 -> stats2 -> final scores),
// avoiding 400MB of materialized activations. GIN segment-sum uses per-block
// LDS accumulators for the 200 hot VM rows. gfx1250-specific paths:
// v_wmma, s_wait_dscnt, ds_add_f32 LDS atomics, global_prefetch_b8.
#include <hip/hip_runtime.h>
#include <stdint.h>

#define NEGV (-100000000.0f)
#define BN_EPS 1e-5f

typedef __attribute__((ext_vector_type(16))) __bf16    v16bf;
typedef __attribute__((ext_vector_type(8)))  float     v8f;
typedef __attribute__((ext_vector_type(4)))  uint32_t  v4u;

union ABf { v16bf v; uint16_t u[16]; v4u q[2]; };

__device__ __forceinline__ uint16_t f2bfu(float f) {
  union { float f; uint32_t u; } x; x.f = f;
  uint32_t r = x.u + 0x7FFFu + ((x.u >> 16) & 1u);   // round-to-nearest-even
  return (uint16_t)(r >> 16);
}

__device__ __forceinline__ v8f wmma_bf16(v16bf a, v16bf b, v8f c) {
  return __builtin_amdgcn_wmma_f32_16x16x32_bf16(false, a, false, b,
                                                 (short)0, c, false, false);
}

// ---------------------------------------------------------------------------
// Utility kernels
// ---------------------------------------------------------------------------
__global__ void k_zero(float* p, int n) {
  int i = blockIdx.x * blockDim.x + threadIdx.x;
  if (i < n) p[i] = 0.f;
}

__global__ void k_init_out(float* p, int n) {
  int i = blockIdx.x * blockDim.x + threadIdx.x;
  if (i < n) p[i] = NEGV;
}

__global__ void k_make_inputs(const float* __restrict__ tss, const float* __restrict__ tsr,
                              const float* __restrict__ tl,  const float* __restrict__ vct,
                              const float* __restrict__ vs,  const float* __restrict__ ver,
                              int T, int V, float* __restrict__ out) {
  int i = blockIdx.x * blockDim.x + threadIdx.x;
  int N = T + V;
  if (i < N) {
    if (i < T) { out[i*3+0] = tss[i]; out[i*3+1] = tsr[i]; out[i*3+2] = tl[i]; }
    else { int v = i - T; out[i*3+0] = vct[v]; out[i*3+1] = vs[v]; out[i*3+2] = ver[v]; }
  }
}

__global__ void k_build_edges(const int* __restrict__ compat, const int* __restrict__ deps,
                              int EC, int ED, int T,
                              int* __restrict__ esrc, int* __restrict__ edst) {
  int i = blockIdx.x * blockDim.x + threadIdx.x;
  int E = EC + ED;
  if (i < E) {
    if (i < EC) { esrc[i] = compat[i]; edst[i] = compat[EC + i] + T; }
    else { int j = i - EC; esrc[i] = deps[j]; edst[i] = deps[ED + j]; }
  }
}

__global__ void k_to_bf16(const float* __restrict__ x, int n, uint16_t* __restrict__ o) {
  int i = blockIdx.x * blockDim.x + threadIdx.x;
  if (i < n) o[i] = f2bfu(x[i]);
}

__global__ void k_colsum(const float* __restrict__ x, int n, float* __restrict__ out32) {
  __shared__ float s[32];
  if (threadIdx.x < 32) s[threadIdx.x] = 0.f;
  __syncthreads();
  int i = blockIdx.x * blockDim.x + threadIdx.x;
  if (i < n) atomicAdd(&s[i & 31], x[i]);
  __syncthreads();
  if (threadIdx.x < 32) atomicAdd(&out32[threadIdx.x], s[threadIdx.x]);
}

__global__ void k_graph_fin(const float* __restrict__ sum, float invn, float* __restrict__ g) {
  int i = threadIdx.x;
  if (i < 32) g[i] = sum[i] * invn;
}

// stats layout: sum at [0..dout), sumsq at [64..64+dout)
__global__ void k_bn_fin(const float* __restrict__ stats, const float* __restrict__ g,
                         const float* __restrict__ be, float invn, int dout,
                         float* __restrict__ a, float* __restrict__ c) {
  int i = threadIdx.x;
  if (i < dout) {
    float m = stats[i] * invn;
    float v = stats[64 + i] * invn - m * m;
    float s = g[i] / sqrtf(v + BN_EPS);
    a[i] = s;
    c[i] = be[i] - s * m;
  }
}

// ---------------------------------------------------------------------------
// Generic fused row-MLP layer (node-level: rows*32 work, VALU is right here).
// If stats != null: OUT holds raw z (pre-BN/relu of NEXT stage), and the
// per-feature sum/sumsq of z are accumulated (block LDS -> global atomics).
// ---------------------------------------------------------------------------
__global__ void k_row_mlp(const float* __restrict__ X, const float* __restrict__ AGG,
                          const float* __restrict__ BNa, const float* __restrict__ BNc,
                          int inRelu,
                          const float* __restrict__ W, const float* __restrict__ Bv,
                          int din, int dout, int rows,
                          float* __restrict__ OUT, int outRelu,
                          float* __restrict__ stats) {
  __shared__ float sstat[128];
  const bool hasStats = (stats != nullptr);
  if (hasStats) for (int i = threadIdx.x; i < 128; i += blockDim.x) sstat[i] = 0.f;
  __syncthreads();
  int r = blockIdx.x * blockDim.x + threadIdx.x;
  if (r < rows) {
    float x[32];
    for (int i = 0; i < din; ++i) {
      float v = X[(size_t)r * din + i];
      if (AGG) v += AGG[(size_t)r * din + i];
      if (BNa) v = BNa[i] * v + BNc[i];
      if (inRelu) v = fmaxf(v, 0.f);
      x[i] = v;
    }
    for (int o = 0; o < dout; ++o) {
      const float* wr = W + (size_t)o * din;
      float acc = Bv[o];
      for (int i = 0; i < din; ++i) acc = fmaf(x[i], wr[i], acc);
      if (outRelu) acc = fmaxf(acc, 0.f);
      OUT[(size_t)r * dout + o] = acc;
      if (hasStats) { atomicAdd(&sstat[o], acc); atomicAdd(&sstat[64 + o], acc * acc); }
    }
  }
  if (hasStats) {
    __syncthreads();
    for (int i = threadIdx.x; i < 128; i += blockDim.x) atomicAdd(&stats[i], sstat[i]);
  }
}

// ---------------------------------------------------------------------------
// GIN segment-sum. Compat edges: 1M edges -> 200 VM rows => block-local LDS
// accumulator (V*32 floats, ds_add_f32) + per-block partials + reduction.
// ---------------------------------------------------------------------------
__global__ void k_seg_compat(const float* __restrict__ h, const int* __restrict__ compat,
                             int EC, int V, float* __restrict__ part) {
  extern __shared__ float vacc[];   // V*32
  int tid = threadIdx.x;
  for (int i = tid; i < V * 32; i += blockDim.x) vacc[i] = 0.f;
  __syncthreads();
  int stride = gridDim.x * blockDim.x;
  for (int e = blockIdx.x * blockDim.x + tid; e < EC; e += stride) {
    int t = compat[e], v = compat[EC + e];
    const float* hr = h + (size_t)t * 32;
    #pragma unroll
    for (int f = 0; f < 32; ++f) atomicAdd(&vacc[v * 32 + f], hr[f]);
  }
  __syncthreads();
  float* pb = part + (size_t)blockIdx.x * V * 32;
  for (int i = tid; i < V * 32; i += blockDim.x) pb[i] = vacc[i];
}

__global__ void k_reduce_vm(const float* __restrict__ part, int nblk, int V,
                            float* __restrict__ aggvm) {
  int i = blockIdx.x * blockDim.x + threadIdx.x;
  if (i < V * 32) {
    float s = 0.f;
    for (int b = 0; b < nblk; ++b) s += part[(size_t)b * V * 32 + i];
    aggvm[i] += s;
  }
}

__global__ void k_seg_deps(const float* __restrict__ h, const int* __restrict__ deps,
                           int ED, float* __restrict__ agg) {
  int i = blockIdx.x * blockDim.x + threadIdx.x;
  if (i < ED * 32) {
    int e = i >> 5, f = i & 31;
    int s = deps[e], d = deps[ED + e];
    atomicAdd(&agg[(size_t)d * 32 + f], h[(size_t)s * 32 + f]);
  }
}

// ---------------------------------------------------------------------------
// Edge scorer: WMMA bf16 passes. Each wave owns 16-edge tiles.
// PASS 0: layer1 -> stats of z1 (64 feats)
// PASS 1: layer1+BN1+relu -> layer2 -> stats of z2 (32 feats)
// PASS 2: full -> scores
// ---------------------------------------------------------------------------
template<int PASS>
__global__ __launch_bounds__(256) void k_edge_pass(
    const uint16_t* __restrict__ node_bf, const float* __restrict__ gemb,
    const int* __restrict__ esrc, const int* __restrict__ edst,
    const float* __restrict__ W1, const float* __restrict__ B1,
    const float* __restrict__ A1, const float* __restrict__ C1,
    const float* __restrict__ W2, const float* __restrict__ B2,
    const float* __restrict__ A2, const float* __restrict__ C2,
    const float* __restrict__ W3, const float* __restrict__ B3,
    float* __restrict__ scores, float* __restrict__ stats,
    int ntiles, int nedges) {
  __shared__ float sstat[128];
  __shared__ float stile[8][16 * 64];   // per-wave 16x64 f32 staging (transpose C->A)

  const int tid  = threadIdx.x;
  const int lane = tid & 31;
  const int wv   = tid >> 5;
  const int m    = lane & 15;
  const int hf   = lane >> 4;           // wave32 half (A/B register layout split)

  if (PASS < 2) for (int i = tid; i < 128; i += 256) sstat[i] = 0.f;
  __syncthreads();

  // Pack B-layout bf16 weight tiles once per wave (amortized over ~13 tiles).
  // B 32x16 bf16: lane<16 -> N=lane,K=kb*32+{0..7,16..23} ; lane>=16 -> N=lane-16,K=+{8..15,24..31}
  v16bf b1t[3][4];
  #pragma unroll
  for (int kb = 0; kb < 3; ++kb)
    #pragma unroll
    for (int nb = 0; nb < 4; ++nb) {
      const float* wr = W1 + (size_t)(nb * 16 + m) * 96 + kb * 32 + hf * 8;
      ABf t;
      #pragma unroll
      for (int j = 0; j < 16; ++j) t.u[j] = f2bfu(wr[j + (j & 8)]);
      b1t[kb][nb] = t.v;
    }
  v16bf b2t[2][2];
  if constexpr (PASS >= 1) {
    #pragma unroll
    for (int kb = 0; kb < 2; ++kb)
      #pragma unroll
      for (int nb = 0; nb < 2; ++nb) {
        const float* wr = W2 + (size_t)(nb * 16 + m) * 64 + kb * 32 + hf * 8;
        ABf t;
        #pragma unroll
        for (int j = 0; j < 16; ++j) t.u[j] = f2bfu(wr[j + (j & 8)]);
        b2t[kb][nb] = t.v;
      }
  }
  // Graph-embedding A block (identical for every edge row).
  // A 16x32 bf16: lane<16 -> row=lane, K={0..7,16..23} ; lane>=16 -> row=lane-16, K={8..15,24..31}
  v16bf a_g;
  {
    ABf t; int b = hf * 8;
    #pragma unroll
    for (int j = 0; j < 8; ++j) { t.u[j] = f2bfu(gemb[b + j]); t.u[8 + j] = f2bfu(gemb[b + 16 + j]); }
    a_g = t.v;
  }

  float b1v[4], a1v[4] = {0,0,0,0}, c1v[4] = {0,0,0,0};
  #pragma unroll
  for (int nb = 0; nb < 4; ++nb) {
    b1v[nb] = B1[nb * 16 + m];
    if constexpr (PASS >= 1) { a1v[nb] = A1[nb * 16 + m]; c1v[nb] = C1[nb * 16 + m]; }
  }
  float b2v[2] = {0,0}, a2v[2] = {0,0}, c2v[2] = {0,0};
  if constexpr (PASS >= 1) {
    #pragma unroll
    for (int nb = 0; nb < 2; ++nb) {
      b2v[nb] = B2[nb * 16 + m];
      if constexpr (PASS == 2) { a2v[nb] = A2[nb * 16 + m]; c2v[nb] = C2[nb * 16 + m]; }
    }
  }

  float s0[4] = {0,0,0,0}, q0[4] = {0,0,0,0};
  float s1[2] = {0,0},     q1[2] = {0,0};

  const int nwaves = gridDim.x * 8;
  for (int t = blockIdx.x * 8 + wv; t < ntiles; t += nwaves) {
    int e = t * 16 + m;
    if (e >= nedges) e = nedges - 1;
    int rs = esrc[e];
    int rd = edst[e];
    __builtin_prefetch(esrc + e + nwaves * 16, 0, 0);   // global_prefetch_b8

    // A tiles: gather bf16 node rows (2 x 16B per lane, L2-resident)
    const uint16_t* rowS = node_bf + (size_t)rs * 32;
    const uint16_t* rowD = node_bf + (size_t)rd * 32;
    ABf ts_, td_;
    int boff = hf * 8;
    ts_.q[0] = *(const v4u*)(rowS + boff);
    ts_.q[1] = *(const v4u*)(rowS + boff + 16);
    td_.q[0] = *(const v4u*)(rowD + boff);
    td_.q[1] = *(const v4u*)(rowD + boff + 16);
    v16bf a_s = ts_.v, a_d = td_.v;

    // layer 1: (16x96) @ (96x64) -> 12 WMMAs, K chained through C
    v8f cc[4];
    #pragma unroll
    for (int nb = 0; nb < 4; ++nb) {
      v8f c = {0.f,0.f,0.f,0.f,0.f,0.f,0.f,0.f};
      c = wmma_bf16(a_s, b1t[0][nb], c);
      c = wmma_bf16(a_d, b1t[1][nb], c);
      c = wmma_bf16(a_g, b1t[2][nb], c);
      cc[nb] = c;
    }

    if constexpr (PASS == 0) {
      #pragma unroll
      for (int nb = 0; nb < 4; ++nb)
        #pragma unroll
        for (int r = 0; r < 8; ++r) {
          float z = cc[nb][r] + b1v[nb];
          s0[nb] += z; q0[nb] += z * z;
        }
    } else {
      // BN1 + relu, C-layout -> LDS row-major 16x64 (transpose for A of layer2)
      float* tile = stile[wv];
      #pragma unroll
      for (int nb = 0; nb < 4; ++nb)
        #pragma unroll
        for (int r = 0; r < 8; ++r) {
          float z = cc[nb][r] + b1v[nb];
          float y = fmaxf(a1v[nb] * z + c1v[nb], 0.f);
          tile[(r + 8 * hf) * 64 + nb * 16 + m] = y;
        }
      asm volatile("s_wait_dscnt 0" ::: "memory");

      const float* hr = tile + m * 64;
      v16bf a2k[2];
      #pragma unroll
      for (int kb = 0; kb < 2; ++kb) {
        ABf t; int b = kb * 32 + hf * 8;
        #pragma unroll
        for (int j = 0; j < 8; ++j) { t.u[j] = f2bfu(hr[b + j]); t.u[8 + j] = f2bfu(hr[b + 16 + j]); }
        a2k[kb] = t.v;
      }

      // layer 2: (16x64) @ (64x32) -> 4 WMMAs
      v8f dd[2];
      #pragma unroll
      for (int nb = 0; nb < 2; ++nb) {
        v8f c = {0.f,0.f,0.f,0.f,0.f,0.f,0.f,0.f};
        c = wmma_bf16(a2k[0], b2t[0][nb], c);
        c = wmma_bf16(a2k[1], b2t[1][nb], c);
        dd[nb] = c;
      }

      if constexpr (PASS == 1) {
        #pragma unroll
        for (int nb = 0; nb < 2; ++nb)
          #pragma unroll
          for (int r = 0; r < 8; ++r) {
            float z = dd[nb][r] + b2v[nb];
            s1[nb] += z; q1[nb] += z * z;
          }
      } else {
        // BN2 + relu -> LDS 16x32, then layer3 dot with w3
        #pragma unroll
        for (int nb = 0; nb < 2; ++nb)
          #pragma unroll
          for (int r = 0; r < 8; ++r) {
            float z = dd[nb][r] + b2v[nb];
            float y = fmaxf(a2v[nb] * z + c2v[nb], 0.f);
            tile[(r + 8 * hf) * 32 + nb * 16 + m] = y;
          }
        asm volatile("s_wait_dscnt 0" ::: "memory");
        if (lane < 16) {
          float acc = B3[0];
          const float* h2 = tile + lane * 32;
          #pragma unroll
          for (int j = 0; j < 32; ++j) acc = fmaf(h2[j], W3[j], acc);
          int eo = t * 16 + lane;
          if (eo < nedges) scores[eo] = acc;
        }
      }
    }
  }

  // Flush stats: lane pair combine (wave32 halves share the feature index),
  // then block LDS, then global atomics.
  if constexpr (PASS == 0) {
    #pragma unroll
    for (int nb = 0; nb < 4; ++nb) {
      s0[nb] += __shfl_xor(s0[nb], 16, 32);
      q0[nb] += __shfl_xor(q0[nb], 16, 32);
    }
    if (lane < 16) {
      #pragma unroll
      for (int nb = 0; nb < 4; ++nb) {
        atomicAdd(&sstat[nb * 16 + lane], s0[nb]);
        atomicAdd(&sstat[64 + nb * 16 + lane], q0[nb]);
      }
    }
  }
  if constexpr (PASS == 1) {
    #pragma unroll
    for (int nb = 0; nb < 2; ++nb) {
      s1[nb] += __shfl_xor(s1[nb], 16, 32);
      q1[nb] += __shfl_xor(q1[nb], 16, 32);
    }
    if (lane < 16) {
      #pragma unroll
      for (int nb = 0; nb < 2; ++nb) {
        atomicAdd(&sstat[nb * 16 + lane], s1[nb]);
        atomicAdd(&sstat[64 + nb * 16 + lane], q1[nb]);
      }
    }
  }
  if constexpr (PASS < 2) {
    __syncthreads();
    for (int i = tid; i < 128; i += 256) atomicAdd(&stats[i], sstat[i]);
  }
}

// ---------------------------------------------------------------------------
// Output assembly
// ---------------------------------------------------------------------------
__global__ void k_scatter(const float* __restrict__ scores, const int* __restrict__ compat,
                          const float* __restrict__ tsr, int EC, int V, float* __restrict__ out) {
  int e = blockIdx.x * blockDim.x + threadIdx.x;
  if (e < EC) {
    int t = compat[e], v = compat[EC + e];
    if (tsr[t] != 0.f) out[(size_t)t * V + v] = scores[e];
  }
}

// ---------------------------------------------------------------------------
// Host orchestration
// ---------------------------------------------------------------------------
extern "C" void kernel_launch(void* const* d_in, const int* in_sizes, int n_in,
                              void* d_out, int out_size, void* d_ws, size_t ws_size,
                              hipStream_t stream) {
  const float* tss = (const float*)d_in[0];
  const float* tsr = (const float*)d_in[1];
  const float* tl  = (const float*)d_in[2];
  const float* vct = (const float*)d_in[3];
  const float* vs  = (const float*)d_in[4];
  const float* ver = (const float*)d_in[5];
  const int* compat = (const int*)d_in[6];
  const int* deps   = (const int*)d_in[7];

  const int T = in_sizes[0];
  const int V = in_sizes[3];
  const int EC = in_sizes[6] / 2;
  const int ED = in_sizes[7] / 2;
  const int N = T + V;
  const int E = EC + ED;

  auto F = [&](int i) { return (const float*)d_in[i]; };
  // pytree flatten order: dict keys sorted.
  // te/ve/es: b1,b2,b3,be1,be2,g1,g2,w1,w2,w3 ; gin layer: b1,b2,be1,g1,w1,w2
  struct EncP { const float *b1,*b2,*b3,*be1,*be2,*g1,*g2,*w1,*w2,*w3; };
  EncP te = { F(8), F(9), F(10), F(11), F(12), F(13), F(14), F(15), F(16), F(17) };
  EncP ve = { F(18),F(19),F(20), F(21), F(22), F(23), F(24), F(25), F(26), F(27) };
  struct GinP { const float *b1,*b2,*be1,*g1,*w1,*w2; };
  GinP gin[3];
  for (int L = 0; L < 3; ++L) {
    int b = 28 + L * 6;
    gin[L] = { F(b+0), F(b+1), F(b+2), F(b+3), F(b+4), F(b+5) };
  }
  EncP es = { F(46),F(47),F(48), F(49), F(50), F(51), F(52), F(53), F(54), F(55) };

  // workspace carve-up
  char* base = (char*)d_ws;
  size_t off = 0;
  auto take = [&](size_t bytes) -> char* {
    char* p = base + off;
    off += (bytes + 255) & ~(size_t)255;
    return p;
  };
  const int SEGB = 128;   // blocks for compat segment-sum
  float*    node_in = (float*)   take((size_t)N * 3 * 4);
  float*    hbuf    = (float*)   take((size_t)N * 32 * 4);
  float*    zbuf    = (float*)   take((size_t)N * 32 * 4);
  float*    tbuf    = (float*)   take((size_t)N * 32 * 4);
  float*    agg     = (float*)   take((size_t)N * 32 * 4);
  uint16_t* node_bf = (uint16_t*)take((size_t)N * 32 * 2);
  int*      esrc    = (int*)     take((size_t)E * 4);
  int*      edst    = (int*)     take((size_t)E * 4);
  float*    scores  = (float*)   take((size_t)E * 4);
  float*    vm_part = (float*)   take((size_t)SEGB * V * 32 * 4);
  float*    stats   = (float*)   take(1024 * 4);
  float* st0  = stats;         // [0:128) generic sum|sumsq
  float* aG   = stats + 128;   // node-level BN scale
  float* cG   = stats + 192;   // node-level BN shift
  float* gsum = stats + 256;
  float* gemb = stats + 288;
  float* a1e  = stats + 320;   // edge BN1 (64)
  float* c1e  = stats + 384;
  float* a2e  = stats + 448;   // edge BN2 (32)
  float* c2e  = stats + 480;

  const int TB = 256;
  auto nb = [](long n) { return (int)((n + 255) / 256); };

  k_build_edges<<<nb(E), TB, 0, stream>>>(compat, deps, EC, ED, T, esrc, edst);
  k_make_inputs<<<nb(N), TB, 0, stream>>>(tss, tsr, tl, vct, vs, ver, T, V, node_in);

  // encoders (BN batch over T rows for tasks, V rows for VMs)
  auto runEnc = [&](const EncP& p, int row0, int rows, float invn) {
    const float* in = node_in + (size_t)row0 * 3;
    float* z1 = zbuf + (size_t)row0 * 32;
    float* z2 = tbuf + (size_t)row0 * 32;
    float* ot = hbuf + (size_t)row0 * 32;
    k_zero<<<1, 128, 0, stream>>>(st0, 128);
    k_row_mlp<<<nb(rows), TB, 0, stream>>>(in, nullptr, nullptr, nullptr, 0,
                                           p.w1, p.b1, 3, 32, rows, z1, 0, st0);
    k_bn_fin<<<1, 64, 0, stream>>>(st0, p.g1, p.be1, invn, 32, aG, cG);
    k_zero<<<1, 128, 0, stream>>>(st0, 128);
    k_row_mlp<<<nb(rows), TB, 0, stream>>>(z1, nullptr, aG, cG, 1,
                                           p.w2, p.b2, 32, 32, rows, z2, 0, st0);
    k_bn_fin<<<1, 64, 0, stream>>>(st0, p.g2, p.be2, invn, 32, aG, cG);
    k_row_mlp<<<nb(rows), TB, 0, stream>>>(z2, nullptr, aG, cG, 1,
                                           p.w3, p.b3, 32, 32, rows, ot, 0, nullptr);
  };
  runEnc(te, 0, T, 1.f / (float)T);
  runEnc(ve, T, V, 1.f / (float)V);

  // GIN x3
  for (int L = 0; L < 3; ++L) {
    k_zero<<<nb((long)N * 32), TB, 0, stream>>>(agg, N * 32);
    k_seg_compat<<<SEGB, TB, (size_t)V * 32 * 4, stream>>>(hbuf, compat, EC, V, vm_part);
    k_reduce_vm<<<nb((long)V * 32), TB, 0, stream>>>(vm_part, SEGB, V, agg + (size_t)T * 32);
    k_seg_deps<<<nb((long)ED * 32), TB, 0, stream>>>(hbuf, deps, ED, agg);
    k_zero<<<1, 128, 0, stream>>>(st0, 128);
    k_row_mlp<<<nb(N), TB, 0, stream>>>(hbuf, agg, nullptr, nullptr, 0,
                                        gin[L].w1, gin[L].b1, 32, 32, N, zbuf, 0, st0);
    k_bn_fin<<<1, 64, 0, stream>>>(st0, gin[L].g1, gin[L].be1, 1.f / (float)N, 32, aG, cG);
    k_row_mlp<<<nb(N), TB, 0, stream>>>(zbuf, nullptr, aG, cG, 1,
                                        gin[L].w2, gin[L].b2, 32, 32, N, hbuf,
                                        (L < 2) ? 1 : 0, nullptr);
  }

  // graph mean + bf16 node embeddings
  k_zero<<<1, 32, 0, stream>>>(gsum, 32);
  k_colsum<<<nb((long)N * 32), TB, 0, stream>>>(hbuf, N * 32, gsum);
  k_graph_fin<<<1, 32, 0, stream>>>(gsum, 1.f / (float)N, gemb);
  k_to_bf16<<<nb((long)N * 32), TB, 0, stream>>>(hbuf, N * 32, node_bf);

  // edge scorer: 3 WMMA passes (stats1, stats2, final)
  const int ntiles = (E + 15) / 16;
  const int EGRID = 640;
  k_zero<<<1, 128, 0, stream>>>(st0, 128);
  k_edge_pass<0><<<EGRID, TB, 0, stream>>>(node_bf, gemb, esrc, edst,
      es.w1, es.b1, nullptr, nullptr, nullptr, nullptr, nullptr, nullptr,
      nullptr, nullptr, nullptr, st0, ntiles, E);
  k_bn_fin<<<1, 64, 0, stream>>>(st0, es.g1, es.be1, 1.f / (float)E, 64, a1e, c1e);
  k_zero<<<1, 128, 0, stream>>>(st0, 128);
  k_edge_pass<1><<<EGRID, TB, 0, stream>>>(node_bf, gemb, esrc, edst,
      es.w1, es.b1, a1e, c1e, es.w2, es.b2, nullptr, nullptr,
      nullptr, nullptr, nullptr, st0, ntiles, E);
  k_bn_fin<<<1, 64, 0, stream>>>(st0, es.g2, es.be2, 1.f / (float)E, 32, a2e, c2e);
  k_edge_pass<2><<<EGRID, TB, 0, stream>>>(node_bf, gemb, esrc, edst,
      es.w1, es.b1, a1e, c1e, es.w2, es.b2, a2e, c2e,
      es.w3, es.b3, scores, nullptr, ntiles, E);

  // output: NEG-init then masked scatter of compat scores
  float* out = (float*)d_out;
  k_init_out<<<nb((long)T * V), TB, 0, stream>>>(out, T * V);
  k_scatter<<<nb(EC), TB, 0, stream>>>(scores, compat, tsr, EC, V, out);
}
